// SCNN_47648367181864
// MI455X (gfx1250) — matrix operand order repaired and checked by reading, not compiled, not run on hardware
//
#include <hip/hip_runtime.h>
#include <math.h>

typedef __attribute__((ext_vector_type(16))) _Float16 v16h;
typedef __attribute__((ext_vector_type(8)))  float    v8f;

// ======================================================================
// Implicit-GEMM convolution on the CDNA5 WMMA path.
//   D[M,N] = im2col(X)[M,K] x W[K,N]  (f16 inputs, f32 accumulate)
// Workgroup = 128 threads (4 waves) owning a 16(M) x 64(N) tile; wave w
// computes the 16x16 sub-tile at N offset w*16. Per 32-wide K chunk the
// block stages the shared 16x32 A fragment into LDS in WMMA fragment order
// (contiguous k: slot t of lane L holds k=(L>>4)*16+t, row m=L&15),
// so each lane reads 32 bytes of LDS per chunk (2x ds_load_b128). Weights
// are pre-converted to f16, so the B fragment is 16 consecutive halves per
// lane: one 32-byte aligned load (2x global_load_b128), zero VALU.
// Templated on window (3x3 / 1x1 / 1x9): constant-divisor k decomposition.
// REQUIREMENT: K multiple of 32 (true for every call). OOB M rows / N cols
// are clamped to 0 (loads in-bounds, stores guarded); WMMA rows/cols are
// independent so valid outputs are unaffected.
// Epilogue modes:
//   0: y = acc*a + c                      (a,c from BN and/or bias; else 1,0)
//   1: y = relu(acc*a + c)
//   2: y = relu(acc*a + c + Res)          (residual add, same layout as out)
//   3: y = Res + relu(acc*a + c)          (SCNN combine, Res == Y in-place)
// ======================================================================
template<int KH, int KW>
__global__ __launch_bounds__(128)
void k_conv_gemm_wmma(const float* __restrict__ X, const _Float16* __restrict__ Wt,
                      float* __restrict__ Y, const float* __restrict__ Res,
                      const float* __restrict__ bng, const float* __restrict__ bnb,
                      const float* __restrict__ bnm, const float* __restrict__ bnv,
                      const float* __restrict__ bias, int mode,
                      int M, int N, int K,
                      int H, int W, int OH, int OW, int stride, int padh, int padw,
                      long isB, long isC, long isY, long isX,
                      long osB, long osC, long osY, long osX)
{
  constexpr int KHKW = KH * KW;
  __shared__ _Float16 As[512] __attribute__((aligned(32)));  // [lane32][slot16]

  const int tid  = threadIdx.x;       // 0..127
  const int lane = tid & 31;          // lane within wave (EXEC all 1s)
  const int wv   = tid >> 5;          // wave id 0..3 -> N sub-tile
  const int hh   = lane >> 4;
  const int l16  = lane & 15;
  const int ohow = OH * OW;

  // ---- A staging role: thread stages fragment entries e = tid*4 .. +3 ----
  // entry e: fragment lane fl=e>>4, slot t=e&15. Slots t0..t0+3 cover 4
  // consecutive k values: k = (fl>>4)*16 + t0 + q.
  const int fl  = tid >> 2;           // fragment lane 0..31
  const int t0  = (tid & 3) * 4;      // 0,4,8,12
  const int sm  = blockIdx.x * 16 + (fl & 15);
  int ab, aoy, aox;
  { int r = (sm < M) ? sm : 0; ab = r / ohow; r -= ab * ohow; aoy = r / OW; aox = r - aoy * OW; }
  const int  biy   = aoy * stride - padh;
  const int  bix   = aox * stride - padw;
  const long abase = (long)ab * isB;
  const int  kfb   = (fl >> 4) * 16 + t0;

  // ---- B column for this lane/wave; clamp OOB cols to col 0 ----
  const int bn   = (blockIdx.y * 4 + wv) * 16 + l16;
  const bool bnok = bn < N;
  const _Float16* wrow = Wt + (long)(bnok ? bn : 0) * K;  // f16 weights [N,K]

  v8f acc = {0.f,0.f,0.f,0.f,0.f,0.f,0.f,0.f};

  for (int kk = 0; kk < K; kk += 32) {
    // stage 4 consecutive-k A elements (im2col gather, pad -> 0, cvt f16)
#pragma unroll
    for (int q = 0; q < 4; ++q) {
      const int k = kk + kfb + q;
      int ci, rr, ss;
      if (KHKW == 1) { ci = k; rr = 0; ss = 0; }
      else {
        ci = k / KHKW;                       // constant divisor
        const int rem = k - ci * KHKW;
        rr = rem / KW;  ss = rem - rr * KW;  // constant divisors
      }
      const int iy = biy + rr, ix = bix + ss;
      const bool ok = (iy >= 0) & (iy < H) & (ix >= 0) & (ix < W);
      const long off = abase + (long)ci*isC + (long)iy*isY + (long)ix*isX;
      const float av = X[ok ? off : 0];      // unconditional in-bounds load
      As[fl * 16 + t0 + q] = (_Float16)(ok ? av : 0.f);
    }
    __syncthreads();
    // A fragment: 32 contiguous bytes of LDS per lane (2 x ds_load_b128)
    const v16h va = *reinterpret_cast<const v16h*>(&As[lane * 16]);
    // B fragment: 16 consecutive f16 per lane, 32B aligned (2 x b128, no VALU)
    const v16h vb = *reinterpret_cast<const v16h*>(wrow + kk + hh * 16);
    __syncthreads();                          // reads done; LDS reusable
    acc = __builtin_amdgcn_wmma_f32_16x16x32_f16(false, va, false, vb,
                                                 (short)0, acc, false, false);
  }

  // ---- fused epilogue: per-channel affine (BN and/or bias) + combine ----
  float affa = 1.f, affc = 0.f;
  if (bnok) {
    if (bng) { const float ia = bng[bn] * rsqrtf(bnv[bn] + 1e-5f);
               affa = ia; affc = bnb[bn] - bnm[bn] * ia; }
    if (bias) affc += bias[bn] * affa;       // bn(conv+bias) fold; a==1 if no BN
  }
#pragma unroll
  for (int vq = 0; vq < 8; ++vq) {
    const int row = blockIdx.x * 16 + vq + 8*hh;   // D layout: VGPR v -> M=v+8*half
    if (row < M && bnok) {
      int r = row; const int ob = r / ohow; r -= ob * ohow;
      const int oy = r / OW, ox = r - oy * OW;
      const long oi = (long)ob*osB + (long)bn*osC + (long)oy*osY + (long)ox*osX;
      float y = acc[vq] * affa + affc;
      if      (mode == 1) y = fmaxf(y, 0.f);
      else if (mode == 2) y = fmaxf(y + Res[oi], 0.f);
      else if (mode == 3) y = Res[oi] + fmaxf(y, 0.f);
      Y[oi] = y;
    }
  }
}

// one-shot weight conversion f32 -> f16 (weights constant for the pass)
__global__ void k_cvt_f16(const float* __restrict__ X, _Float16* __restrict__ Y, int n)
{
  int i = blockIdx.x * blockDim.x + threadIdx.x;
  if (i < n) Y[i] = (_Float16)X[i];
}

// ======================================================================
// conv1: 7x7 s2 p3, Cin=3 (too thin for WMMA K-tiling) + BN + ReLU. Direct.
// ======================================================================
__global__ void k_conv7_bn_relu(const float* __restrict__ X, const float* __restrict__ W7,
                                const float* __restrict__ g, const float* __restrict__ b,
                                const float* __restrict__ m, const float* __restrict__ v,
                                float* __restrict__ Y, int total)
{
  int i = blockIdx.x * blockDim.x + threadIdx.x;
  if (i >= total) return;
  int ox = i % 400; int t = i / 400;
  int oy = t % 144; t /= 144;
  int co = t % 64;  int nb = t / 64;
  float acc = 0.f;
  const float* wp = W7 + (long)co * 3 * 49;
  for (int ci = 0; ci < 3; ++ci)
    for (int r = 0; r < 7; ++r) {
      int iy = oy*2 - 3 + r;
      if (iy < 0 || iy >= 288) continue;
      const float* xr = X + ((long)(nb*3 + ci)*288 + iy)*800;
      for (int s = 0; s < 7; ++s) {
        int ix = ox*2 - 3 + s;
        if (ix >= 0 && ix < 800) acc += xr[ix] * wp[(ci*7 + r)*7 + s];
      }
    }
  float a = g[co] * rsqrtf(v[co] + 1e-5f);
  float c = b[co] - m[co]*a;
  Y[i] = fmaxf(acc*a + c, 0.f);
}

// maxpool 3x3 s2 p1: [8,64,144,400] -> [8,64,72,200]
__global__ void k_maxpool(const float* __restrict__ X, float* __restrict__ Y, int total)
{
  int i = blockIdx.x * blockDim.x + threadIdx.x;
  if (i >= total) return;
  int ox = i % 200; int t = i / 200;
  int oy = t % 72;  t /= 72;
  int c  = t % 64;  int nb = t / 64;
  float mx = -1e30f;
  const float* p = X + ((long)(nb*64 + c)*144)*400;
  for (int r = 0; r < 3; ++r) {
    int iy = oy*2 - 1 + r;
    if (iy < 0 || iy >= 144) continue;
    for (int s = 0; s < 3; ++s) {
      int ix = ox*2 - 1 + s;
      if (ix >= 0 && ix < 400) mx = fmaxf(mx, p[iy*400 + ix]);
    }
  }
  Y[i] = mx;
}

// bilinear align_corners resize: [8,5,9,25] -> [8,5,144,400]
__global__ void k_resize(const float* __restrict__ X, float* __restrict__ Y, int total)
{
  int i = blockIdx.x * blockDim.x + threadIdx.x;
  if (i >= total) return;
  int ox = i % 400; int t = i / 400;
  int oy = t % 144; t /= 144;
  int c  = t % 5;   int nb = t / 5;
  float fy = oy * (8.0f / 143.0f);
  int y0 = (int)floorf(fy); int y1 = (y0 + 1 > 8) ? 8 : y0 + 1;
  float wy = fy - (float)y0;
  float fx = ox * (24.0f / 399.0f);
  int x0 = (int)floorf(fx); int x1 = (x0 + 1 > 24) ? 24 : x0 + 1;
  float wx = fx - (float)x0;
  const float* p = X + ((long)(nb*5 + c)*9)*25;
  float v00 = p[y0*25 + x0], v10 = p[y1*25 + x0];
  float v01 = p[y0*25 + x1], v11 = p[y1*25 + x1];
  Y[i] = (v00*(1.f-wy) + v10*wy)*(1.f-wx) + (v01*(1.f-wy) + v11*wy)*wx;
}

// global average pool: [8,512,9,25] -> [8,512]
__global__ void k_mean(const float* __restrict__ X, float* __restrict__ Y, int total)
{
  int i = blockIdx.x * blockDim.x + threadIdx.x;
  if (i >= total) return;
  const float* p = X + (long)i * 225;
  float s = 0.f;
  for (int t = 0; t < 225; ++t) s += p[t];
  Y[i] = s * (1.0f / 225.0f);
}

// dense: Y[b,o] = (relu?)(X[b,:] . W[o,:] + bias[o])
__global__ void k_fc(const float* __restrict__ X, const float* __restrict__ Wt,
                     const float* __restrict__ bias, float* __restrict__ Y,
                     int B, int I, int O, int relu)
{
  int i = blockIdx.x * blockDim.x + threadIdx.x;
  if (i >= B*O) return;
  int o = i % O, nb = i / O;
  float acc = bias[o];
  const float* x = X + (long)nb * I;
  const float* w = Wt + (long)o * I;
  for (int k = 0; k < I; ++k) acc += x[k] * w[k];
  Y[i] = relu ? fmaxf(acc, 0.f) : acc;
}

// ======================================================================
extern "C" void kernel_launch(void* const* d_in, const int* in_sizes, int n_in,
                              void* d_out, int out_size, void* d_ws, size_t ws_size,
                              hipStream_t stream)
{
  (void)in_sizes; (void)n_in; (void)out_size; (void)ws_size;
  auto f = [&](int i) { return (const float*)d_in[i]; };

  struct BN { const float *b, *g, *m, *v; };
  int idx = 0;
  const float* Xin = f(idx++);                                  // x
  auto getBN = [&]() { BN p; p.b=f(idx++); p.g=f(idx++); p.m=f(idx++); p.v=f(idx++); return p; };
  BN bn1 = getBN();                                             // bn1 (b,g,m,v)
  const float* w_conv1 = f(idx++);                              // conv1
  const float* ex_b1 = f(idx++); const float* ex_b2 = f(idx++); // ex_b1, ex_b2
  const float* ex_w1 = f(idx++); const float* ex_w2 = f(idx++); // ex_w1, ex_w2
  struct Blk { BN bn1, bn2, bnd; const float *w1, *w2, *wd; bool down; };
  auto getBlk = [&](bool down) {
    Blk k = {}; k.down = down;
    k.bn1 = getBN(); k.bn2 = getBN(); if (down) k.bnd = getBN();
    k.w1 = f(idx++); k.w2 = f(idx++); k.wd = down ? f(idx++) : nullptr;
    return k;
  };
  Blk blks[8] = { getBlk(false), getBlk(false),   // layer1
                  getBlk(true),  getBlk(false),   // layer2
                  getBlk(true),  getBlk(false),   // layer3
                  getBlk(true),  getBlk(false) }; // layer4
  const float* scnn_d = f(idx++); const float* scnn_l = f(idx++);
  const float* scnn_r = f(idx++); const float* scnn_u = f(idx++);
  const float* seg_b1 = f(idx++); const float* seg_b2 = f(idx++);
  BN seg_bn = getBN();
  const float* seg_w1 = f(idx++); const float* seg_w2 = f(idx++);

  // ---- workspace layout (floats) ----
  float* ws  = (float*)d_ws;
  const long SLOT = 7372800;            // 8*64*72*200 (largest post-pool map)
  float* bufA = ws;                     // 4*SLOT: conv1 out, later scratch
  float* bufB = ws + 4*SLOT;            // running feature map (ping)
  float* P1   = bufA;                   // t1
  float* P2   = bufA + SLOT;            // downsample shortcut
  float* P3   = bufA + 2*SLOT;          // pong
  float* SEGa = bufA + 3*SLOT;          // 8*128*9*25 = 230400
  float* SEGb = SEGa + 230400;          // 8*5*9*25   = 9000
  float* MEANb = SEGb + 16384;          // 8*512
  float* FC1b  = MEANb + 4096;          // 8*128
  // f16 weight arena starts after bufB (5*SLOT floats in): ~20.7M halves
  _Float16* wcur = (_Float16*)(ws + 5*SLOT);

  // one-shot f32->f16 weight conversion into the arena (sizes all %32==0)
  auto cvtw = [&](const float* w, long n) -> const _Float16* {
    _Float16* dst = wcur; wcur += n;
    k_cvt_f16<<<dim3((unsigned)((n + 255)/256)), 256, 0, stream>>>(w, dst, (int)n);
    return dst;
  };
  struct BlkH { const _Float16 *w1, *w2, *wd; };
  const long wsz[4][3] = { {  36864L*4/4,  36864L,      0L },   // per-layer sizes
                           {  73728L,     147456L,  8192L },
                           { 294912L,     589824L, 32768L },
                           {1179648L,    2359296L,131072L } };
  BlkH blkh[8];
  for (int li = 0; li < 4; ++li) {
    long n1 = wsz[li][0], n2 = wsz[li][1], nd = wsz[li][2];
    // first block of layer
    blkh[li*2+0].w1 = cvtw(blks[li*2+0].w1, n1);
    blkh[li*2+0].w2 = cvtw(blks[li*2+0].w2, n2);
    blkh[li*2+0].wd = blks[li*2+0].down ? cvtw(blks[li*2+0].wd, nd) : nullptr;
    // second block: both convs are Cout x Cout x 3 x 3 = n2
    blkh[li*2+1].w1 = cvtw(blks[li*2+1].w1, n2);
    blkh[li*2+1].w2 = cvtw(blks[li*2+1].w2, n2);
    blkh[li*2+1].wd = nullptr;
  }
  const _Float16* scnn_dh = cvtw(scnn_d, 2359296L);
  const _Float16* scnn_uh = cvtw(scnn_u, 2359296L);
  const _Float16* scnn_rh = cvtw(scnn_r, 2359296L);
  const _Float16* scnn_lh = cvtw(scnn_l, 2359296L);
  const _Float16* seg_w1h = cvtw(seg_w1, 65536L);
  const _Float16* seg_w2h = cvtw(seg_w2, 640L);

  auto gemm = [&](const float* Xp, const _Float16* Wp, float* Yp, const float* Rp,
                  const BN* bnp, const float* biasp, int mode,
                  int Bn, int Cin, int Cout, int H, int W, int kh, int kw,
                  int stride, int padh, int padw,
                  long isB_, long isC_, long isY_, long isX_,
                  int OH, int OW, long osB_, long osC_, long osY_, long osX_) {
    int M = Bn*OH*OW, N = Cout, K = Cin*kh*kw;   // K is a multiple of 32 here
    dim3 grid((M + 15)/16, (N + 63)/64);         // 4 waves/block: 16M x 64N tile
    const float* g_ = bnp ? bnp->g : nullptr;
    const float* b_ = bnp ? bnp->b : nullptr;
    const float* m_ = bnp ? bnp->m : nullptr;
    const float* v_ = bnp ? bnp->v : nullptr;
#define LAUNCH_GEMM(KHV, KWV)                                               \
    k_conv_gemm_wmma<KHV, KWV><<<grid, 128, 0, stream>>>(Xp, Wp, Yp, Rp,    \
        g_, b_, m_, v_, biasp, mode, M, N, K,                               \
        H, W, OH, OW, stride, padh, padw,                                   \
        isB_, isC_, isY_, isX_, osB_, osC_, osY_, osX_)
    if      (kh == 3 && kw == 3) LAUNCH_GEMM(3, 3);
    else if (kh == 1 && kw == 9) LAUNCH_GEMM(1, 9);
    else                         LAUNCH_GEMM(1, 1);
#undef LAUNCH_GEMM
  };

  // ---- stem ----
  { int total = 8*64*144*400;
    k_conv7_bn_relu<<<(total+255)/256, 256, 0, stream>>>(
        Xin, w_conv1, bn1.g, bn1.b, bn1.m, bn1.v, bufA, total); }
  { int total = 8*64*72*200;
    k_maxpool<<<(total+255)/256, 256, 0, stream>>>(bufA, bufB, total); }

  // ---- residual blocks (WMMA implicit GEMM, fused BN/residual/ReLU) ----
  float* cur = bufB;
  int C = 64, H = 72, W = 200;
  const int cfg[4][2] = {{64,1},{128,2},{256,2},{512,2}};
  for (int li = 0; li < 4; ++li) {
    for (int bi = 0; bi < 2; ++bi) {
      const Blk&  bk = blks[li*2 + bi];
      const BlkH& bh = blkh[li*2 + bi];
      int Cout = cfg[li][0];
      int s = (bi == 0) ? cfg[li][1] : 1;
      int OH = (H + 2 - 3)/s + 1, OW = (W + 2 - 3)/s + 1;
      float* out = (cur == bufB) ? P3 : bufB;
      // t1 = relu(bn1(conv3x3(cur)))
      gemm(cur, bh.w1, P1, nullptr, &bk.bn1, nullptr, 1,
           8, C, Cout, H, W, 3, 3, s, 1, 1,
           (long)C*H*W, (long)H*W, W, 1,
           OH, OW, (long)Cout*OH*OW, (long)OH*OW, OW, 1);
      const float* sc = cur;
      if (bk.down) {           // sc = bnd(conv1x1(cur))
        gemm(cur, bh.wd, P2, nullptr, &bk.bnd, nullptr, 0,
             8, C, Cout, H, W, 1, 1, s, 0, 0,
             (long)C*H*W, (long)H*W, W, 1,
             OH, OW, (long)Cout*OH*OW, (long)OH*OW, OW, 1);
        sc = P2;
      }
      // out = relu(bn2(conv3x3(t1)) + sc)
      gemm(P1, bh.w2, out, sc, &bk.bn2, nullptr, 2,
           8, Cout, Cout, OH, OW, 3, 3, 1, 1, 1,
           (long)Cout*OH*OW, (long)OH*OW, OW, 1,
           OH, OW, (long)Cout*OH*OW, (long)OH*OW, OW, 1);
      cur = out; C = Cout; H = OH; W = OW;
    }
  }
  // cur == bufB, [8,512,9,25]

  // ---- SCNN directional passes (sequential line GEMMs, in-place) ----
  const long sc_ = 225, sb_ = 512L*225;
  for (int i = 1; i < 9; ++i)      // down: conv along W (25), line stride 25
    gemm(cur + (long)(i-1)*25, scnn_dh, cur + (long)i*25, cur + (long)i*25,
         nullptr, nullptr, 3, 8, 512, 512, 1, 25, 1, 9, 1, 0, 4,
         sb_, sc_, 25, 1, 1, 25, sb_, sc_, 25, 1);
  for (int i = 7; i >= 0; --i)     // up
    gemm(cur + (long)(i+1)*25, scnn_uh, cur + (long)i*25, cur + (long)i*25,
         nullptr, nullptr, 3, 8, 512, 512, 1, 25, 1, 9, 1, 0, 4,
         sb_, sc_, 25, 1, 1, 25, sb_, sc_, 25, 1);
  for (int j = 1; j < 25; ++j)     // right: conv along H (9), element stride 25
    gemm(cur + (j-1), scnn_rh, cur + j, cur + j,
         nullptr, nullptr, 3, 8, 512, 512, 1, 9, 1, 9, 1, 0, 4,
         sb_, sc_, 1, 25, 1, 9, sb_, sc_, 1, 25);
  for (int j = 23; j >= 0; --j)    // left
    gemm(cur + (j+1), scnn_lh, cur + j, cur + j,
         nullptr, nullptr, 3, 8, 512, 512, 1, 9, 1, 9, 1, 0, 4,
         sb_, sc_, 1, 25, 1, 9, sb_, sc_, 1, 25);

  // ---- seg head: 1x1 convs via WMMA (bias folded into BN affine) ----
  gemm(cur, seg_w1h, SEGa, nullptr, &seg_bn, seg_b1, 1,
       8, 512, 128, 9, 25, 1, 1, 1, 0, 0,
       sb_, sc_, 25, 1, 9, 25, 128L*225, 225, 25, 1);
  gemm(SEGa, seg_w2h, SEGb, nullptr, nullptr, seg_b2, 0,
       8, 128, 5, 9, 25, 1, 1, 1, 0, 0,
       128L*225, 225, 25, 1, 9, 25, 5L*225, 225, 25, 1);

  float* out = (float*)d_out;
  { int total = 8*5*144*400;
    k_resize<<<(total+255)/256, 256, 0, stream>>>(SEGb, out, total); }

  // ---- existence head ----
  { int total = 8*512;
    k_mean<<<(total+255)/256, 256, 0, stream>>>(cur, MEANb, total); }
  k_fc<<<(8*128+255)/256, 256, 0, stream>>>(MEANb, ex_w1, ex_b1, FC1b, 8, 512, 128, 1);
  k_fc<<<1, 64, 0, stream>>>(FC1b, ex_w2, ex_b2, out + 2304000, 8, 128, 4, 0);
}
